// RNNgain_2095944040493
// MI455X (gfx1250) — matrix operand order repaired and checked by reading, not compiled, not run on hardware
//
#include <hip/hip_runtime.h>
#include <hip/hip_bf16.h>
#include <stdint.h>

#define B_  64
#define T_  1024
#define N_  1024
#define TN_ (1024*1024)
#define ALPHA     0.05f
#define NOISE_STD 0.01f
#define NWG 32            // workgroups in persistent RNN kernel; each owns 32 columns

typedef __attribute__((ext_vector_type(16))) __bf16 v16bf;
typedef __attribute__((ext_vector_type(8)))  float  v8f;

#if defined(__has_builtin)
#if __has_builtin(__builtin_amdgcn_tensor_load_to_lds)
#define HAVE_TDM 1
#endif
#endif

#ifdef HAVE_TDM
typedef unsigned int u32x4 __attribute__((ext_vector_type(4)));
typedef int          i32x4 __attribute__((ext_vector_type(4)));
typedef int          i32x8 __attribute__((ext_vector_type(8)));
#endif

__device__ __forceinline__ unsigned short f2bf(float x) {
    unsigned u = __float_as_uint(x);
    unsigned r = u + 0x7fffu + ((u >> 16) & 1u);   // round-to-nearest-even
    return (unsigned short)(r >> 16);
}

// softplus(x) = max(x,0) + ln(1 + 2^(-log2e*|x|)) via hardware v_exp_f32 / v_log_f32:
// exp2 arg <= 0 and log arg in [1,2], so no libm range-reduction sequence is needed.
__device__ __forceinline__ float softplusf(float x) {
    float e = __builtin_amdgcn_exp2f(-1.4426950408889634f * fabsf(x));
    return fmaxf(x, 0.f) + 0.6931471805599453f * __builtin_amdgcn_logf(1.f + e);
}

// ---------------- kernel 0: zero out + reset barrier/y0 (runs every launch -> graph-replay safe)
__global__ void k_zero(float* __restrict__ out, unsigned* __restrict__ bar, float* __restrict__ y0) {
    int i = blockIdx.x * 256 + threadIdx.x;
    if (i < B_ * T_) out[i] = 0.f;
    if (i == 0) { bar[0] = 0u; *y0 = 0.f; }
}

// ---------------- kernel 1: Wg[j][k] = relu(wrec*refEI)*refEI*mwrec * gpos[k], bf16, row-major by j
__global__ void k_weights(const float* __restrict__ wrec, const float* __restrict__ refEI,
                          const float* __restrict__ mwrec, const float* __restrict__ g,
                          unsigned short* __restrict__ Wg) {
    int i = blockIdx.x * 256 + threadIdx.x;       // 0 .. N*N-1
    int k = i & (N_ - 1);
    float e  = wrec[i] * refEI[i];
    e = fmaxf(e, 0.f) * refEI[i] * mwrec[i];
    e *= fmaxf(g[k], 0.f);
    Wg[i] = f2bf(e);
}

// ---------------- kernel 2: r0 = softplus(h0+b) broadcast over batch; y0 = sum_j r0[j]*wout[j]
__global__ void k_init_r(const float* __restrict__ h0, const float* __restrict__ bvec,
                         const float* __restrict__ wout,
                         unsigned short* __restrict__ rb0, float* __restrict__ y0) {
    __shared__ float sred[256];
    int j = blockIdx.x * 256 + threadIdx.x;       // 1024 threads total (4 blocks)
    float r = softplusf(h0[j] + bvec[j]);
    unsigned short rbv = f2bf(r);
    for (int b = 0; b < B_; ++b) rb0[b * N_ + j] = rbv;
    sred[threadIdx.x] = r * wout[j];
    __syncthreads();
    for (int s = 128; s > 0; s >>= 1) {
        if (threadIdx.x < s) sred[threadIdx.x] += sred[threadIdx.x + s];
        __syncthreads();
    }
    if (threadIdx.x == 0) atomicAdd(y0, sred[0]);
}

// ---------------- main persistent RNN kernel: 32 WGs x 256 threads (8 waves)
// wave w: mtile = w&3 (16 batch rows), ntile = w>>2 (16 of this WG's 32 columns)
__global__ void __launch_bounds__(256, 1)
k_rnn(const float* __restrict__ input, const float* __restrict__ noise,
      const float* __restrict__ wi,    const float* __restrict__ h0,
      const float* __restrict__ wout,  const float* __restrict__ bvec,
      const float* __restrict__ g,
      const unsigned short* __restrict__ Wg,
      unsigned short* __restrict__ rb0v, unsigned short* __restrict__ rb1v,
      unsigned* __restrict__ bar, const float* __restrict__ y0,
      float* __restrict__ out)
{
    __shared__ unsigned short lw[32 * 1024];   // 64KB: this WG's K x 32-col weight slice (j-major)

    const int tid = threadIdx.x;
    const int wg  = blockIdx.x;
    const int j0  = wg * 32;

    if (wg == 0 && tid < B_) out[(size_t)tid * T_] = *y0;   // out[:,0] (same for all b)

    // ---- stage weight slice into LDS: Tensor Data Mover (async DMA), fallback = vector copy
#ifdef HAVE_TDM
    if (tid < 32) {   // wave 0 issues one TDM descriptor: 64KB contiguous tile, 4B elements
        unsigned long long ga = (unsigned long long)(uintptr_t)(Wg + (size_t)j0 * N_);
        u32x4 g0;
        g0[0] = 1u;                                   // count=1, user mode
        g0[1] = (unsigned)(uintptr_t)&lw[0];          // lds_addr (byte offset)
        g0[2] = (unsigned)ga;                         // global_addr[31:0]
        g0[3] = (unsigned)((ga >> 32) & 0x01FFFFFFull) | 0x80000000u;  // addr[56:32] | type=2
        i32x8 g1;
        g1[0] = 0x00020000;                           // data_size=2 (4B), no mask/flags
        g1[1] = (int)(0x4000u << 16);                 // tensor_dim0 = 16384 (lo16 in [31:16])
        g1[2] = (int)(1u << 16);                      // tensor_dim0 hi=0; tensor_dim1 = 1
        g1[3] = (int)(0x4000u << 16);                 // tensor_dim1 hi=0; tile_dim0 = 16384
        g1[4] = 1;                                    // tile_dim1 = 1; tile_dim2 = 0
        g1[5] = 0x4000;                               // tensor_dim0_stride = 16384
        g1[6] = 0;
        g1[7] = 0;
        i32x4 gz4 = {0, 0, 0, 0};
        i32x8 gz8 = {0, 0, 0, 0, 0, 0, 0, 0};
        __builtin_amdgcn_tensor_load_to_lds(g0, g1, gz4, gz4, gz8, 0);
        __builtin_amdgcn_s_wait_tensorcnt(0);
    }
#else
    {
        const uint4* src = (const uint4*)(Wg + (size_t)j0 * N_);
        uint4* dst = (uint4*)lw;
        for (int i = tid; i < 4096; i += 256) dst[i] = src[i];
    }
#endif
    __syncthreads();

    const int lane  = tid & 31;
    const int w     = tid >> 5;
    const int mtile = w & 3;
    const int ntile = w >> 2;
    const int nIdx  = lane & 15;
    const int hi    = lane >> 4;
    const int jloc  = ntile * 16 + nIdx;
    const int j     = j0 + jloc;
    const int bbase = mtile * 16;

    const float bT_j  = bvec[j];
    const float wi0_j = wi[j];
    const float wi1_j = wi[N_ + j];
    const float gw_j  = fmaxf(g[j], 0.f) * wout[j];
    const float h0_j  = h0[j];

    float h[8];
    #pragma unroll
    for (int v = 0; v < 8; ++v) h[v] = h0_j;

    // A-fragment base: row = bbase+nIdx; lanes<16 take K {0..15}, lanes>=16 take {16..31}
    const size_t arow = (size_t)(bbase + nIdx) * N_ + (hi ? 16 : 0);
    // B-fragment base in LDS: col jloc, 16 contiguous K at hi*16 within each 32-chunk
    const int    boff = jloc * 1024 + (hi ? 16 : 0);

    // per-row constant indices (index-based addressing keeps loads in the GLOBAL path,
    // with the inter-row deltas folded into the 24-bit instruction offsets)
    size_t nidx[8];   // into noise (f32 elements)
    int    iidx[8];   // into input viewed as float2
    int    roff[8];   // into r ping-pong buffers (bf16 elements)
    size_t oidx[8];   // into out (f32 elements)
    #pragma unroll
    for (int v = 0; v < 8; ++v) {
        const int bg = bbase + v + hi * 8;
        nidx[v] = (size_t)bg * TN_ + j;
        iidx[v] = bg * T_;
        roff[v] = bg * N_ + j;
        oidx[v] = (size_t)bg * T_;
    }

    const float2* inp2 = (const float2*)input;   // (B,T,2) -> float2[b*T + t]
    size_t tN = 0;                                // t * N_

    for (int t = 0; t < T_ - 1; ++t) {
        const unsigned short* rc = (t & 1) ? rb1v : rb0v;
        unsigned short*       rn = (t & 1) ? rb0v : rb1v;

        v8f acc0 = {0.f, 0.f, 0.f, 0.f, 0.f, 0.f, 0.f, 0.f};
        v8f acc1 = {0.f, 0.f, 0.f, 0.f, 0.f, 0.f, 0.f, 0.f};
        const unsigned short* aptr = rc + arow;

        // dual accumulators break the WMMA->WMMA RAW chain (pipelined matrix ops)
        #pragma unroll 2
        for (int kc = 0; kc < 32; kc += 2) {
            union { uint4 u[2]; v16bf v; } a0, a1, b0, b1;
            a0.u[0] = *(const uint4*)(aptr + kc * 32);
            a0.u[1] = *(const uint4*)(aptr + kc * 32 + 8);
            a1.u[0] = *(const uint4*)(aptr + kc * 32 + 32);
            a1.u[1] = *(const uint4*)(aptr + kc * 32 + 40);
            b0.u[0] = *(const uint4*)(lw + boff + kc * 32);
            b0.u[1] = *(const uint4*)(lw + boff + kc * 32 + 8);
            b1.u[0] = *(const uint4*)(lw + boff + kc * 32 + 32);
            b1.u[1] = *(const uint4*)(lw + boff + kc * 32 + 40);
            acc0 = __builtin_amdgcn_wmma_f32_16x16x32_bf16(false, a0.v, false, b0.v,
                                                           (short)0, acc0, false, false);
            acc1 = __builtin_amdgcn_wmma_f32_16x16x32_bf16(false, a1.v, false, b1.v,
                                                           (short)0, acc1, false, false);
        }

        // recurrence update; acc/h live in the WMMA C/D layout: (M = v + 8*hi, N = nIdx)
        float yp[8];
        #pragma unroll
        for (int v = 0; v < 8; ++v) {
            float  nz = noise[nidx[v] + tN];
            __builtin_prefetch(noise + nidx[v] + tN + N_, 0, 3);  // next step's noise (HBM stream)
            float2 u  = inp2[iidx[v] + t];
            float hv  = h[v];
            hv = hv + NOISE_STD * nz
               + ALPHA * (-hv + (acc0[v] + acc1[v]) + u.x * wi0_j + u.y * wi1_j);
            h[v] = hv;
            float r = softplusf(hv + bT_j);
            rn[roff[v]] = f2bf(r);
            yp[v] = r * gw_j;
        }

        // y[b, t+1] += sum_j r*gw : reduce over the 16 lanes of each half, then atomic per row
        #pragma unroll
        for (int v = 0; v < 8; ++v) {
            float red = yp[v];
            red += __shfl_xor(red, 1, 32);
            red += __shfl_xor(red, 2, 32);
            red += __shfl_xor(red, 4, 32);
            red += __shfl_xor(red, 8, 32);
            if (nIdx == 0)
                atomicAdd(out + oidx[v] + (t + 1), red);
        }

        tN += N_;

        // grid-wide barrier (skip after the final step: last r is never consumed)
        if (t < T_ - 2) {
            __threadfence();
            __syncthreads();
            if (tid == 0) {
                __hip_atomic_fetch_add(bar, 1u, __ATOMIC_RELEASE, __HIP_MEMORY_SCOPE_AGENT);
                const unsigned target = (unsigned)(t + 1) * NWG;
                while (__hip_atomic_load(bar, __ATOMIC_ACQUIRE, __HIP_MEMORY_SCOPE_AGENT) < target)
                    __builtin_amdgcn_s_sleep(1);
            }
            __syncthreads();
            __threadfence();
        }
    }
}

extern "C" void kernel_launch(void* const* d_in, const int* in_sizes, int n_in,
                              void* d_out, int out_size, void* d_ws, size_t ws_size,
                              hipStream_t stream) {
    const float* input = (const float*)d_in[0];
    const float* noise = (const float*)d_in[1];
    const float* wi    = (const float*)d_in[2];
    const float* wrec  = (const float*)d_in[3];
    const float* mwrec = (const float*)d_in[4];
    const float* refEI = (const float*)d_in[5];
    const float* h0    = (const float*)d_in[6];
    const float* wout  = (const float*)d_in[7];
    const float* bvec  = (const float*)d_in[8];
    const float* g     = (const float*)d_in[9];
    float* out = (float*)d_out;

    char* ws = (char*)d_ws;
    unsigned short* Wg  = (unsigned short*)ws;                             // 2 MB
    unsigned short* rb0 = (unsigned short*)(ws + (size_t)2 * 1024 * 1024); // 128 KB
    unsigned short* rb1 = rb0 + B_ * N_;                                   // 128 KB
    float*    y0  = (float*)(rb1 + B_ * N_);
    unsigned* bar = (unsigned*)(y0 + 1);

    k_zero   <<<256,  256, 0, stream>>>(out, bar, y0);
    k_weights<<<4096, 256, 0, stream>>>(wrec, refEI, mwrec, g, Wg);
    k_init_r <<<4,    256, 0, stream>>>(h0, bvec, wout, rb0, y0);
    k_rnn    <<<NWG,  256, 0, stream>>>(input, noise, wi, h0, wout, bvec, g,
                                        Wg, rb0, rb1, bar, y0, out);
}
